// YOLOv8Loss_89670327205963
// MI455X (gfx1250) — matrix operand order repaired and checked by reading, not compiled, not run on hardware
//
#include <hip/hip_runtime.h>
#include <math.h>

typedef _Float16 v16h __attribute__((ext_vector_type(16)));
typedef float    v8f  __attribute__((ext_vector_type(8)));

#define NCLS 80
#define NT   50

__device__ __forceinline__ float bce_logits(float x, float y) {
    // max(x,0) - x*y + log1p(exp(-|x|))
    return fmaxf(x, 0.f) - x * y + log1pf(expf(-fabsf(x)));
}

__device__ __forceinline__ float wave_sum(float v) {
#pragma unroll
    for (int off = 16; off > 0; off >>= 1) v += __shfl_xor(v, off, 32);
    return v;
}

__global__ void zero_ws_kernel(float* acc) {
    if (threadIdx.x < 16) acc[threadIdx.x] = 0.f;
}

__global__ __launch_bounds__(256) void yolo_level_kernel(
    const float* __restrict__ pred,     // (B, 4+C, H, W)
    const float* __restrict__ targets,  // (B, T, 5) : cls,cx,cy,w,h
    float* __restrict__ acc,            // [n, sum_iou, sum_cls, sum_obj]
    int W, int H, float stride, int tiles)
{
    __shared__ float s_gxs[NT], s_gys[NT], s_rf[NT];
    __shared__ float s_box[NT][4];
    __shared__ int   s_gi[NT], s_gj[NT], s_r[NT], s_gcls[NT], s_valid[NT];

    const int b  = blockIdx.x;
    const int HW = W * H;
    const int tid = threadIdx.x;

    // ---- per-block target preprocessing into LDS ----
    if (tid < NT) {
        const float* tp = targets + ((size_t)b * NT + tid) * 5;
        float c = tp[0], x = tp[1], y = tp[2], w = tp[3], h = tp[4];
        int valid = ((c + x + y + w + h) != 0.f) ? 1 : 0;
        float gx = x * (float)W, gy = y * (float)H;
        float gw = w * (float)W, gh = h * (float)H;
        int gcls = (int)c; gcls = gcls < 0 ? 0 : (gcls > NCLS - 1 ? NCLS - 1 : gcls);
        int r = (int)(fmaxf(gw, gh) / stride); if (r < 3) r = 3;
        float gxs = gx / stride, gys = gy / stride;
        int gi = (int)gxs; gi = gi < 0 ? 0 : (gi > W - 1 ? W - 1 : gi);
        int gj = (int)gys; gj = gj < 0 ? 0 : (gj > H - 1 ? H - 1 : gj);
        s_gxs[tid] = gxs; s_gys[tid] = gys; s_rf[tid] = (float)r;
        s_gi[tid] = gi; s_gj[tid] = gj; s_r[tid] = r;
        s_gcls[tid] = gcls; s_valid[tid] = valid;
        s_box[tid][0] = gx; s_box[tid][1] = gy; s_box[tid][2] = gw; s_box[tid][3] = gh;
    }
    __syncthreads();

    const int lane = tid & 31;
    const int wave = tid >> 5;
    const int tileIdx = blockIdx.y * 8 + wave;
    if (tileIdx >= tiles) return;            // wave-uniform exit

    const int cellBase = tileIdx * 16;
    const int row16 = lane & 15;             // row of the tile this lane owns
    const int g = lane >> 4;                 // half-wave id (A/B K-group)
    const int myCell = cellBase + row16;
    const int ci = myCell % W;
    const int cj = myCell / W;

    // ---- phase 1: match bitmask over 50 targets for this lane's cell ----
    unsigned mlo = 0u, mhi = 0u;
    int tind = -1;
#pragma unroll 1
    for (int t = 0; t < NT; ++t) {
        int dx_i = ci - s_gi[t]; dx_i = dx_i < 0 ? -dx_i : dx_i;
        int dy_i = cj - s_gj[t]; dy_i = dy_i < 0 ? -dy_i : dy_i;
        float dx = (float)ci - s_gxs[t];
        float dy = (float)cj - s_gys[t];
        float rf = s_rf[t];
        float q = 1.f - (dy * dy + dx * dx) / (2.f * rf * rf);
        int m = s_valid[t] & (dx_i <= s_r[t]) & (dy_i <= s_r[t]) & (q > 0.f);
        if (t < 32) mlo |= ((unsigned)m) << t;
        else        mhi |= ((unsigned)m) << (t - 32);
        tind = m ? t : tind;
    }
    const int anym = ((mlo | mhi) != 0u) ? 1 : 0;
    const unsigned rowmask = (unsigned)__ballot(anym) & 0xFFFFu; // bit r = mask of cell cellBase+r

    // ---- build A (match matrix, 16x64 f16, two K=32 chunks) ----
    // CDNA5 16-bit A layout: element e of v16h holds K = e + 8*g + (e>=8 ? 8 : 0)
    v16h A0, A1;
#pragma unroll
    for (int e = 0; e < 16; ++e) {
        int K = e + 8 * g + ((e >= 8) ? 8 : 0);
        A0[e] = ((mlo >> K) & 1u) ? (_Float16)1.0f : (_Float16)0.0f;
        A1[e] = ((mhi >> K) & 1u) ? (_Float16)1.0f : (_Float16)0.0f; // t=K+32; bits>=18 are 0
    }

    const size_t pbase = (size_t)b * (4 + NCLS) * HW;
    float sum_cls = 0.f;
    float objmax[8];
#pragma unroll
    for (int i = 0; i < 8; ++i) objmax[i] = -__builtin_inff();

    const int ncol = lane & 15;              // class column within the 16-wide tile

    // ---- 5 class tiles: cnt = match(16x64) x onehot(64x16) via WMMA ----
#pragma unroll 1
    for (int ct = 0; ct < 5; ++ct) {
        const int cls = ct * 16 + ncol;
        // CDNA5 16-bit B layout (matches A): element e of v16h holds K = e + 8*g + (e>=8 ? 8 : 0)
        v16h B0, B1;
#pragma unroll
        for (int e = 0; e < 16; ++e) {
            int k0 = e + 8 * g + ((e >= 8) ? 8 : 0);
            int k1 = k0 + 32;
            B0[e] = (s_gcls[k0] == cls) ? (_Float16)1.0f : (_Float16)0.0f;
            B1[e] = (k1 < NT && s_gcls[k1] == cls) ? (_Float16)1.0f : (_Float16)0.0f;
        }
        v8f cacc = {0.f, 0.f, 0.f, 0.f, 0.f, 0.f, 0.f, 0.f};
        cacc = __builtin_amdgcn_wmma_f32_16x16x32_f16(false, A0, false, B0,
                                                      (short)0, cacc, false, false);
        cacc = __builtin_amdgcn_wmma_f32_16x16x32_f16(false, A1, false, B1,
                                                      (short)0, cacc, false, false);

        // fused: cls BCE + per-cell class max (for objectness)
        const float* pc_col = pred + pbase + (size_t)(4 + cls) * HW;
#pragma unroll
        for (int i = 0; i < 8; ++i) {
            int rrow = i + 8 * g;            // C/D layout: VGPR i -> row i+8g
            int cell = cellBase + rrow;
            float mcv = (cacc[i] > 0.f) ? 1.f : 0.f;
            float mfv = (float)((rowmask >> rrow) & 1u);
            float pcv = pc_col[cell];
            sum_cls += bce_logits(pcv, mcv) * mfv;
            objmax[i] = fmaxf(objmax[i], pcv);
        }
    }

    // ---- objectness: reduce max across the 16-lane class group, BCE once per cell ----
    float sum_obj = 0.f;
#pragma unroll
    for (int i = 0; i < 8; ++i) {
        float v = objmax[i];
#pragma unroll
        for (int off = 1; off < 16; off <<= 1) v = fmaxf(v, __shfl_xor(v, off, 32));
        int rrow = i + 8 * g;
        float mfv = (float)((rowmask >> rrow) & 1u);
        float sel = (ncol == 0) ? 1.f : 0.f;  // one lane per 16-lane group contributes
        sum_obj += sel * bce_logits(v, mfv);
    }

    // ---- IoU / lbox term (lanes 0-15 each own one cell; 16-31 are duplicates) ----
    float sum_iou = 0.f, cnt_n = 0.f;
    if (lane < 16) {
        float p0v = pred[pbase + 0 * (size_t)HW + myCell];
        float p1v = pred[pbase + 1 * (size_t)HW + myCell];
        float p2v = pred[pbase + 2 * (size_t)HW + myCell];
        float p3v = pred[pbase + 3 * (size_t)HW + myCell];
        float t0 = 0.f, t1 = 0.f, t2 = 0.f, t3 = 0.f;
        if (tind >= 0) {
            t0 = s_box[tind][0]; t1 = s_box[tind][1];
            t2 = s_box[tind][2]; t3 = s_box[tind][3];
        }
        float pa = (p2v - p0v) * (p3v - p1v);
        float ta = (t2 - t0) * (t3 - t1);
        float wi = fminf(p2v, t2) - fmaxf(p0v, t0);
        float hi = fminf(p3v, t3) - fmaxf(p1v, t1);
        float ai = wi * hi;
        float denom = pa + ta - ai;
        if (anym) { sum_iou += ai / denom; cnt_n += 1.f; }
    }

    sum_cls = wave_sum(sum_cls);
    sum_obj = wave_sum(sum_obj);
    sum_iou = wave_sum(sum_iou);
    cnt_n   = wave_sum(cnt_n);
    if (lane == 0) {
        atomicAdd(acc + 0, cnt_n);
        atomicAdd(acc + 1, sum_iou);
        atomicAdd(acc + 2, sum_cls);
        atomicAdd(acc + 3, sum_obj);
    }
}

__global__ void finalize_kernel(const float* __restrict__ acc,
                                float* __restrict__ out, int B) {
    const int HWs[3] = {6400, 1600, 400};
    float lbox = 0.f, lcls = 0.f, lobj = 0.f;
    for (int l = 0; l < 3; ++l) {
        float n    = acc[l * 4 + 0];
        float siou = acc[l * 4 + 1];
        float scls = acc[l * 4 + 2];
        float sobj = acc[l * 4 + 3];
        lbox += (n > 0.f) ? siou / fmaxf(n, 1.f) : 0.f;
        lcls += scls / fmaxf(n * (float)NCLS, 1.f);
        lobj += sobj / ((float)B * (float)HWs[l]);
    }
    lbox *= 5.f;
    float loss = lbox + lcls + lobj;
    out[0] = loss;   // tuple element 0: loss
    out[1] = lbox;   // tuple element 1: stacked stats
    out[2] = lcls;
    out[3] = lobj;
    out[4] = loss;
}

extern "C" void kernel_launch(void* const* d_in, const int* in_sizes, int n_in,
                              void* d_out, int out_size, void* d_ws, size_t ws_size,
                              hipStream_t stream) {
    const float* preds[3] = { (const float*)d_in[0], (const float*)d_in[1],
                              (const float*)d_in[2] };
    const float* tg = (const float*)d_in[3];
    const int B = in_sizes[3] / (NT * 5);     // = 32

    float* acc = (float*)d_ws;                // 12 floats used
    zero_ws_kernel<<<1, 32, 0, stream>>>(acc);

    const int   Ws[3]      = { 80, 40, 20 };
    const float strides[3] = { 8.f, 16.f, 32.f };
    for (int l = 0; l < 3; ++l) {
        int W = Ws[l], H = Ws[l];
        int tiles = (W * H) / 16;             // 400 / 100 / 25
        dim3 grid(B, (tiles + 7) / 8);
        yolo_level_kernel<<<grid, 256, 0, stream>>>(preds[l], tg, acc + l * 4,
                                                    W, H, strides[l], tiles);
    }
    finalize_kernel<<<1, 1, 0, stream>>>(acc, (float*)d_out, B);
}
